// MultiHeadAttentionRoPE_40364102648030
// MI455X (gfx1250) — matrix operand order repaired and checked by reading, not compiled, not run on hardware
//
#include <hip/hip_runtime.h>
#include <hip/hip_bf16.h>

// ---------------------------------------------------------------------------
// MultiHeadAttention + RoPE for MI455X (gfx1250, wave32, WMMA f16->f32)
//
// B=2, L=2048, D=768, H=12, HD=64.
//   (0) f32 -> f16 conversion of query and weights into d_ws
//   (1) fused QKV GEMM + bias + RoPE; 32x64 tile per wave; fully unrolled
//       k-loop with zero-copy ping-pong double buffering of global loads
//   (2) flash attention, one wave per (b,h,16 query rows); S^T = K*Q^T makes
//       softmax lane-local and P^T registers map directly to the WMMA
//       A-matrix layout; K/V double-buffered via pairwise-unrolled steps
//   (3) output projection GEMM + bias (same structure as (1))
//
// Workspace requirement: 18,087,936 f16 elements (~36.2 MB).
// ---------------------------------------------------------------------------

typedef _Float16 h16v __attribute__((ext_vector_type(16)));
typedef _Float16 h8v  __attribute__((ext_vector_type(8)));
typedef float    f8v  __attribute__((ext_vector_type(8)));

constexpr int CB  = 2;
constexpr int CL  = 2048;
constexpr int CD  = 768;
constexpr int CH  = 12;
constexpr int CBL = CB * CL;               // 4096
constexpr size_t N_X = (size_t)CBL * CD;   // 3,145,728
constexpr size_t N_W = (size_t)CD * CD;    //   589,824

static __device__ __forceinline__ f8v wmma_f16(h16v a, h16v b, f8v c) {
  // D = A(16x32 f16) x B(32x16 f16) + C(16x16 f32)
  return __builtin_amdgcn_wmma_f32_16x16x32_f16(
      /*neg_a=*/false, a, /*neg_b=*/false, b,
      /*c_mod=*/(short)0, c, /*reuse_a=*/false, /*reuse_b=*/false);
}

static __device__ __forceinline__ h16v cat8(h8v lo, h8v hi) {
  return __builtin_shufflevector(lo, hi, 0, 1, 2, 3, 4, 5, 6, 7,
                                         8, 9, 10, 11, 12, 13, 14, 15);
}

// ---------------------------------------------------------------------------
// (0) fp32 -> fp16 conversion
// ---------------------------------------------------------------------------
__global__ void cvt_f32_to_f16(const float* __restrict__ src,
                               _Float16* __restrict__ dst, int n) {
  int i = blockIdx.x * blockDim.x + threadIdx.x;
  if (i < n) dst[i] = (_Float16)src[i];
}

// ---------------------------------------------------------------------------
// (1) fused QKV projection + RoPE, 32 rows x 64 cols per wave.
// grid = (CBL/32, H, 3)
// ---------------------------------------------------------------------------
__global__ __launch_bounds__(32) void qkv_rope_kernel(
    const _Float16* __restrict__ Xh,
    const _Float16* __restrict__ Wqh, const _Float16* __restrict__ Wkh,
    const _Float16* __restrict__ Wvh,
    const float* __restrict__ bq, const float* __restrict__ bk,
    const float* __restrict__ bv,
    _Float16* __restrict__ qR, _Float16* __restrict__ kR,
    _Float16* __restrict__ vT) {
  const int lane = threadIdx.x;
  const int g  = lane >> 4;       // lane group (0/1)
  const int ln = lane & 15;
  const int gb = g * 8;
  const int m0  = blockIdx.x * 32;       // rows in flattened (B*L)
  const int h   = blockIdx.y;            // head
  const int mat = blockIdx.z;            // 0=q 1=k 2=v
  const _Float16* W   = (mat == 0) ? Wqh : (mat == 1) ? Wkh : Wvh;
  const float*    bia = (mat == 0) ? bq  : (mat == 1) ? bk  : bv;
  const int n0 = h * 64;

  const _Float16* arow0 = Xh + (size_t)(m0 + ln) * CD;
  const _Float16* arow1 = Xh + (size_t)(m0 + 16 + ln) * CD;

  auto loadA = [&](int kb, h16v (&a)[2]) {
    a[0] = cat8(*(const h8v*)(arow0 + kb + gb),
                *(const h8v*)(arow0 + kb + 16 + gb));
    a[1] = cat8(*(const h8v*)(arow1 + kb + gb),
                *(const h8v*)(arow1 + kb + 16 + gb));
  };
  auto loadB = [&](int kb, h16v (&bt)[4]) {
#pragma unroll
    for (int t = 0; t < 4; ++t)
      bt[t] = cat8(*(const h8v*)(W + (size_t)(n0 + 16 * t + ln) * CD + kb + gb),
                   *(const h8v*)(W + (size_t)(n0 + 16 * t + ln) * CD + kb + 16 + gb));
  };

  f8v acc[2][4] = {};
  h16v aB[2][2], bB[2][4];     // zero-copy ping-pong buffers
  loadA(0, aB[0]);
  loadB(0, bB[0]);
#pragma unroll
  for (int it = 0; it < CD / 32; ++it) {       // fully unrolled: it&1 constant
    const int kb  = it * 32;
    const int cur = it & 1, nxt = cur ^ 1;
    if (kb + 32 < CD) {        // issue next step's loads before this step's MMAs
      loadA(kb + 32, aB[nxt]);
      loadB(kb + 32, bB[nxt]);
    }
#pragma unroll
    for (int t = 0; t < 4; ++t) {
      acc[0][t] = wmma_f16(aB[cur][0], bB[cur][t], acc[0][t]);
      acc[1][t] = wmma_f16(aB[cur][1], bB[cur][t], acc[1][t]);
    }
  }
#pragma unroll
  for (int t = 0; t < 4; ++t) {
    const float bb = bia[n0 + 16 * t + ln];
#pragma unroll
    for (int s = 0; s < 2; ++s)
#pragma unroll
      for (int r = 0; r < 8; ++r) acc[s][t][r] += bb;
  }

  const int b  = m0 >> 11;            // /2048 (tile never crosses batch)
  const int l0 = m0 & (CL - 1);

  if (mat < 2) {
    // RoPE: pair (d, d+32) lives in acc[s][t] / acc[s][t+2] of the SAME lane.
#pragma unroll
    for (int t = 0; t < 2; ++t) {
      const int i = 16 * t + ln;                          // rotary index 0..31
      const float fr = __expf(-(float)i * 0.28782313662425574f); // ln(1e4)/32
#pragma unroll
      for (int s = 0; s < 2; ++s)
#pragma unroll
        for (int r = 0; r < 8; ++r) {
          const int l = l0 + 16 * s + r + 8 * g;
          float sn, cs;
          __sincosf((float)l * fr, &sn, &cs);
          const float x1 = acc[s][t][r], x2 = acc[s][t + 2][r];
          acc[s][t][r]     = x1 * cs - x2 * sn;
          acc[s][t + 2][r] = x2 * cs + x1 * sn;
        }
    }
    _Float16* dst = (mat == 0) ? qR : kR;
    const size_t base = (size_t)(b * CH + h) * CL * 64;
#pragma unroll
    for (int t = 0; t < 4; ++t) {
      const int d = 16 * t + ln;
#pragma unroll
      for (int s = 0; s < 2; ++s)
#pragma unroll
        for (int r = 0; r < 8; ++r) {
          const int l = l0 + 16 * s + r + 8 * g;
          dst[base + (size_t)l * 64 + d] = (_Float16)acc[s][t][r];
        }
    }
  } else {
    // V stored transposed: (b,h,64,L); 8 consecutive l per lane -> 16B store
    const size_t base = (size_t)(b * CH + h) * 64 * CL;
#pragma unroll
    for (int t = 0; t < 4; ++t) {
      const int d = 16 * t + ln;
#pragma unroll
      for (int s = 0; s < 2; ++s) {
        h8v pack;
#pragma unroll
        for (int r = 0; r < 8; ++r) pack[r] = (_Float16)acc[s][t][r];
        *(h8v*)(vT + base + (size_t)d * CL + l0 + 16 * s + 8 * g) = pack;
      }
    }
  }
}

// ---------------------------------------------------------------------------
// (2) flash attention. grid = (L/16, H, B); one wave per 16-query tile.
// Computes S^T = K @ Q^T so each query row's scores are lane-local.
// K/V double-buffered with pairwise-unrolled steps (zero-copy ping-pong).
// ---------------------------------------------------------------------------
__global__ __launch_bounds__(32) void flash_attn_kernel(
    const _Float16* __restrict__ qR, const _Float16* __restrict__ kR,
    const _Float16* __restrict__ vT, _Float16* __restrict__ X2h) {
  const int lane = threadIdx.x;
  const int g  = lane >> 4;
  const int ln = lane & 15;
  const int gb = g * 8;
  const int q0 = blockIdx.x * 16;
  const int h  = blockIdx.y;
  const int b  = blockIdx.z;
  const size_t bh = (size_t)(b * CH + h);
  const _Float16* Qp = qR + bh * CL * 64;
  const _Float16* Kp = kR + bh * CL * 64;
  const _Float16* Vp = vT + bh * 64 * CL;

  // Q as B-matrix chunks (dims c*32 .. c*32+31): two 16B halves (+gb, +16+gb) per lane
  h16v qb[2];
#pragma unroll
  for (int c = 0; c < 2; ++c)
    qb[c] = cat8(*(const h8v*)(Qp + (size_t)(q0 + ln) * 64 + c * 32 + gb),
                 *(const h8v*)(Qp + (size_t)(q0 + ln) * 64 + c * 32 + 16 + gb));

  // kA index = sub*2+c  (sub: key rows +0/+16, c: dim chunk)
  auto loadK = [&](int kb, h16v (&ka)[4]) {
#pragma unroll
    for (int sub = 0; sub < 2; ++sub) {
      const _Float16* krow = Kp + (size_t)(kb + 16 * sub + ln) * 64;
#pragma unroll
      for (int c = 0; c < 2; ++c)
        ka[sub * 2 + c] = cat8(*(const h8v*)(krow + c * 32 + gb),
                               *(const h8v*)(krow + c * 32 + 16 + gb));
    }
  };
  auto loadV = [&](int kb, h16v (&vb)[4]) {
#pragma unroll
    for (int t = 0; t < 4; ++t)
      vb[t] = cat8(*(const h8v*)(Vp + (size_t)(16 * t + ln) * CL + kb + gb),
                   *(const h8v*)(Vp + (size_t)(16 * t + ln) * CL + kb + 16 + gb));
  };

  f8v o[4] = {};
  float m_run = -3.0e38f, l_run = 0.f;
  const int qg = q0 + ln;
  const int nsteps = (q0 + 47) >> 5;   // cover k < q0+16, 32 keys per step

  // one attention step: consume (kC,vC); optionally prefetch into (kN,vN)
  auto step = [&](int kb, bool pre, h16v (&kC)[4], h16v (&vC)[4],
                  h16v (&kN)[4], h16v (&vN)[4]) {
    if (pre) {
      loadK(kb + 32, kN);
      loadV(kb + 32, vN);
    }
    f8v s1 = {}, s2 = {};
#pragma unroll
    for (int c = 0; c < 2; ++c) {
      s1 = wmma_f16(kC[0 * 2 + c], qb[c], s1);
      s2 = wmma_f16(kC[1 * 2 + c], qb[c], s2);
    }
    // scale, causal mask, online softmax (row = lane pair {ln, ln+16})
    float mloc = -3.0e38f;
#pragma unroll
    for (int r = 0; r < 8; ++r) {
      const int k1 = kb + r + 8 * g;
      float v1 = s1[r] * 0.125f;
      if (k1 > qg) v1 = -3.0e38f;
      s1[r] = v1; mloc = fmaxf(mloc, v1);
      const int k2 = kb + 16 + r + 8 * g;
      float v2 = s2[r] * 0.125f;
      if (k2 > qg) v2 = -3.0e38f;
      s2[r] = v2; mloc = fmaxf(mloc, v2);
    }
    mloc = fmaxf(mloc, __shfl_xor(mloc, 16, 32));
    const float m_new = fmaxf(m_run, mloc);
    const float alpha = __expf(m_run - m_new);
    float psum = 0.f;
#pragma unroll
    for (int r = 0; r < 8; ++r) {
      s1[r] = __expf(s1[r] - m_new);
      s2[r] = __expf(s2[r] - m_new);
      psum += s1[r] + s2[r];
    }
    psum += __shfl_xor(psum, 16, 32);
    l_run = l_run * alpha + psum;
    m_run = m_new;

    // P registers already match the WMMA A-matrix half layout:
    // half hv -> v=hv/2, klocal = 2*(v%4)+(hv&1); v<4 -> s1, else s2.
    h16v pa;
#pragma unroll
    for (int hv = 0; hv < 16; ++hv) {
      const int v  = hv >> 1;
      const int kl = 2 * (v & 3) + (hv & 1);
      pa[hv] = (_Float16)((v < 4) ? s1[kl] : s2[kl]);
    }

    // broadcast per-row alpha into the O (lane=dim) layout
    float alp[8];
#pragma unroll
    for (int r = 0; r < 8; ++r) alp[r] = __shfl(alpha, r + 8 * g, 32);
#pragma unroll
    for (int t = 0; t < 4; ++t) {
#pragma unroll
      for (int r = 0; r < 8; ++r) o[t][r] *= alp[r];
      o[t] = wmma_f16(pa, vC[t], o[t]);
    }
  };

  h16v kB0[4], vB0[4], kB1[4], vB1[4];
  loadK(0, kB0);
  loadV(0, vB0);
  int j = 0;
  for (; j + 2 <= nsteps; j += 2) {        // pairwise unroll: constant buffers
    step(j * 32, true, kB0, vB0, kB1, vB1);
    step((j + 1) * 32, j + 2 < nsteps, kB1, vB1, kB0, vB0);
  }
  if (j < nsteps)                           // odd tail
    step(j * 32, false, kB0, vB0, kB1, vB1);

  // normalize and store f16 attention output (flattened (B*L, D))
  float linv[8];
#pragma unroll
  for (int r = 0; r < 8; ++r)
    linv[r] = 1.f / __shfl(l_run, r + 8 * g, 32);
  const int bL = b * CL;
#pragma unroll
  for (int t = 0; t < 4; ++t) {
    const int d = 16 * t + ln;
#pragma unroll
    for (int r = 0; r < 8; ++r) {
      X2h[(size_t)(bL + q0 + r + 8 * g) * CD + h * 64 + d] =
          (_Float16)(o[t][r] * linv[r]);
    }
  }
}

// ---------------------------------------------------------------------------
// (3) output projection: out = X2 @ Wo^T + bo (f32), 32x64 tile per wave.
// grid = (CBL/32, D/64)
// ---------------------------------------------------------------------------
__global__ __launch_bounds__(32) void out_proj_kernel(
    const _Float16* __restrict__ X2h, const _Float16* __restrict__ Woh,
    const float* __restrict__ bo, float* __restrict__ out) {
  const int lane = threadIdx.x;
  const int g  = lane >> 4;
  const int ln = lane & 15;
  const int gb = g * 8;
  const int m0 = blockIdx.x * 32;
  const int n0 = blockIdx.y * 64;

  const _Float16* arow0 = X2h + (size_t)(m0 + ln) * CD;
  const _Float16* arow1 = X2h + (size_t)(m0 + 16 + ln) * CD;

  auto loadA = [&](int kb, h16v (&a)[2]) {
    a[0] = cat8(*(const h8v*)(arow0 + kb + gb),
                *(const h8v*)(arow0 + kb + 16 + gb));
    a[1] = cat8(*(const h8v*)(arow1 + kb + gb),
                *(const h8v*)(arow1 + kb + 16 + gb));
  };
  auto loadB = [&](int kb, h16v (&bt)[4]) {
#pragma unroll
    for (int t = 0; t < 4; ++t)
      bt[t] = cat8(*(const h8v*)(Woh + (size_t)(n0 + 16 * t + ln) * CD + kb + gb),
                   *(const h8v*)(Woh + (size_t)(n0 + 16 * t + ln) * CD + kb + 16 + gb));
  };

  f8v acc[2][4] = {};
  h16v aB[2][2], bB[2][4];
  loadA(0, aB[0]);
  loadB(0, bB[0]);
#pragma unroll
  for (int it = 0; it < CD / 32; ++it) {
    const int kb  = it * 32;
    const int cur = it & 1, nxt = cur ^ 1;
    if (kb + 32 < CD) {
      loadA(kb + 32, aB[nxt]);
      loadB(kb + 32, bB[nxt]);
    }
#pragma unroll
    for (int t = 0; t < 4; ++t) {
      acc[0][t] = wmma_f16(aB[cur][0], bB[cur][t], acc[0][t]);
      acc[1][t] = wmma_f16(aB[cur][1], bB[cur][t], acc[1][t]);
    }
  }
#pragma unroll
  for (int t = 0; t < 4; ++t) {
    const float bb = bo[n0 + 16 * t + ln];
#pragma unroll
    for (int s = 0; s < 2; ++s)
#pragma unroll
      for (int r = 0; r < 8; ++r)
        out[(size_t)(m0 + 16 * s + r + 8 * g) * CD + n0 + 16 * t + ln] =
            acc[s][t][r] + bb;
  }
}

// ---------------------------------------------------------------------------
// host launcher
// ---------------------------------------------------------------------------
extern "C" void kernel_launch(void* const* d_in, const int* in_sizes, int n_in,
                              void* d_out, int out_size, void* d_ws,
                              size_t ws_size, hipStream_t stream) {
  const float* query = (const float*)d_in[0];
  const float* Wq = (const float*)d_in[1];
  const float* bq = (const float*)d_in[2];
  const float* Wk = (const float*)d_in[3];
  const float* bk = (const float*)d_in[4];
  const float* Wv = (const float*)d_in[5];
  const float* bv = (const float*)d_in[6];
  const float* Wo = (const float*)d_in[7];
  const float* bo = (const float*)d_in[8];
  float* out = (float*)d_out;

  _Float16* ws = (_Float16*)d_ws;
  _Float16* Xh  = ws;                    // N_X
  _Float16* Wqh = Xh  + N_X;             // N_W
  _Float16* Wkh = Wqh + N_W;
  _Float16* Wvh = Wkh + N_W;
  _Float16* Woh = Wvh + N_W;
  _Float16* qR  = Woh + N_W;             // (B,H,L,64)
  _Float16* kR  = qR + N_X;              // (B,H,L,64)
  _Float16* vT  = kR + N_X;              // (B,H,64,L)
  _Float16* X2h = vT + N_X;              // (B*L, D)

  // (0) conversions
  {
    const int thr = 256;
    cvt_f32_to_f16<<<(int)((N_X + thr - 1) / thr), thr, 0, stream>>>(query, Xh, (int)N_X);
    cvt_f32_to_f16<<<(int)((N_W + thr - 1) / thr), thr, 0, stream>>>(Wq, Wqh, (int)N_W);
    cvt_f32_to_f16<<<(int)((N_W + thr - 1) / thr), thr, 0, stream>>>(Wk, Wkh, (int)N_W);
    cvt_f32_to_f16<<<(int)((N_W + thr - 1) / thr), thr, 0, stream>>>(Wv, Wvh, (int)N_W);
    cvt_f32_to_f16<<<(int)((N_W + thr - 1) / thr), thr, 0, stream>>>(Wo, Woh, (int)N_W);
  }
  // (1) QKV + RoPE
  qkv_rope_kernel<<<dim3(CBL / 32, CH, 3), 32, 0, stream>>>(
      Xh, Wqh, Wkh, Wvh, bq, bk, bv, qR, kR, vT);
  // (2) flash attention
  flash_attn_kernel<<<dim3(CL / 16, CH, CB), 32, 0, stream>>>(qR, kR, vT, X2h);
  // (3) output projection
  out_proj_kernel<<<dim3(CBL / 32, CD / 64), 32, 0, stream>>>(X2h, Woh, bo, out);
}
